// MultiHeadAttention_5463198401291
// MI455X (gfx1250) — matrix operand order repaired and checked by reading, not compiled, not run on hardware
//
#include <hip/hip_runtime.h>
#include <hip/hip_bf16.h>

typedef __bf16 bf16;
typedef __attribute__((ext_vector_type(16))) __bf16 v16bf;
typedef __attribute__((ext_vector_type(8)))  __bf16 v8bf;
typedef __attribute__((ext_vector_type(8)))  float  v8f;
typedef __attribute__((ext_vector_type(4)))  unsigned int v4u;
typedef __attribute__((ext_vector_type(8)))  int v8i;
typedef __attribute__((ext_vector_type(4)))  int v4i;

#define NB 16      // batch
#define SL 1024    // sequence length
#define EM 1024    // embed size
#define NH 16      // heads
#define HD 64      // head dim

// A-matrix (16x32, 16-bit) per-lane K index for element e (ISA 7.12.2):
// lane: m = lane&15, half = lane>>4; VGPR v=e>>1:
//   k = (v>>2)*16 + half*8 + (v&3)*2 + (e&1)
__device__ __forceinline__ int amap(int e, int half) {
    int v = e >> 1;
    return ((v >> 2) << 4) + (half << 3) + ((v & 3) << 1) + (e & 1);
}

__device__ __forceinline__ v8f wmma_bf16(v16bf a, v16bf b, v8f c) {
    // D = A(16x32) * B(32x16) + C, fp32 accumulate
    return __builtin_amdgcn_wmma_f32_16x16x32_bf16(
        /*neg_a=*/false, a, /*neg_b=*/false, b,
        /*c_mod=*/(short)0, c, /*reuse_a=*/false, /*reuse_b=*/false);
}

// 16 contiguous bf16 (32B, aligned) -> v16bf
__device__ __forceinline__ v16bf load16(const bf16* p) {
    const v8bf* q = (const v8bf*)p;
    v8bf lo = q[0];
    v8bf hi = q[1];
    return __builtin_shufflevector(lo, hi, 0, 1, 2, 3, 4, 5, 6, 7,
                                   8, 9, 10, 11, 12, 13, 14, 15);
}

// ---------------------------------------------------------------------------
// TDM: DMA a 2-D bf16 tile (tile_k rows x tile_n cols, row stride in elems)
// from global memory into LDS.  D# per ISA 8.3/8.4:
//  group0: [1:0]=count=1, [63:32]=lds_addr, [120:64]=global_addr, [127:126]=2
//  group1: data_size=1(2B); tensor_dim0/1 = tile_n/tile_k; tile_dim0/1;
//          tensor_dim0_stride = row_stride.  groups 2/3 = 0 (2-D tensor).
// ---------------------------------------------------------------------------
__device__ __forceinline__ void tdm_load_2d(unsigned lds_off, const void* gptr,
                                            int tile_n, int tile_k, int row_stride) {
    unsigned long long ga = (unsigned long long)(size_t)gptr;
    v4u g0;
    g0[0] = 1u;                                               // count=1, user D#
    g0[1] = lds_off;                                          // LDS byte address
    g0[2] = (unsigned)ga;                                     // global addr lo
    g0[3] = (unsigned)((ga >> 32) & 0x1FFFFFFull) | (2u << 30); // addr hi | type=2
    v8i g1;
    g1[0] = 1 << 16;                                          // data_size = 2B
    g1[1] = (tile_n & 0xFFFF) << 16;                          // tensor_dim0[15:0]
    g1[2] = ((tile_n >> 16) & 0xFFFF) | ((tile_k & 0xFFFF) << 16); // td0 hi | td1 lo
    g1[3] = ((tile_k >> 16) & 0xFFFF) | ((tile_n & 0xFFFF) << 16); // td1 hi | tile_dim0
    g1[4] = tile_k & 0xFFFF;                                  // tile_dim1, tile_dim2=0
    g1[5] = row_stride;                                       // tensor_dim0_stride lo
    g1[6] = 0;
    g1[7] = 0;
    v4i gz = {0, 0, 0, 0};
#if __has_include(<hip/amd_detail/amd_gfx1250_TDM.h>)
    v8i gz8 = {0, 0, 0, 0, 0, 0, 0, 0};
    __builtin_amdgcn_tensor_load_to_lds(g0, g1, gz, gz, gz8, 0);
#else
    __builtin_amdgcn_tensor_load_to_lds(g0, g1, gz, gz, 0);
#endif
}

// ---------------------------------------------------------------------------
// Kernel 1: per-head shared projection  out = x @ W^T, x rows are 64-wide
// slices of (N,L,E); output bf16, optionally transposed (D x L per (n,h)).
// One wave = 16 rows x 64 cols, K=64 (2 chunks of 32), 8 WMMAs.
// ---------------------------------------------------------------------------
__global__ __launch_bounds__(256) void proj_kernel(const float* __restrict__ X,
                                                   const float* __restrict__ W,
                                                   bf16* __restrict__ Out,
                                                   int transposed) {
    const int lane = threadIdx.x & 31;
    const int half = lane >> 4;
    const int ncol = lane & 15;
    const int wv   = threadIdx.x >> 5;
    const int w    = blockIdx.x * 8 + wv;     // 16384 wave tiles
    const int gr0  = w * 16;                  // rows over (n,l,h)

    // A: 16 rows of x (each 64 floats, contiguous), converted to bf16
    const int grA  = gr0 + ncol;
    const int nA   = grA >> 14;
    const int remA = grA & 16383;
    const float* xrow = X + ((size_t)(nA * SL + (remA >> 4)) * EM + (remA & 15) * HD);

    v16bf a[2];
#pragma unroll
    for (int c = 0; c < 2; ++c)
#pragma unroll
        for (int e = 0; e < 16; ++e)
            a[c][e] = (bf16)xrow[c * 32 + amap(e, half)];

    v8f acc[4];
#pragma unroll
    for (int t = 0; t < 4; ++t) acc[t] = (v8f){0.f, 0.f, 0.f, 0.f, 0.f, 0.f, 0.f, 0.f};

    // B[k=d][n=e_out] = W[e_out][d]; B lane = n, elements over k (amap) in W row
#pragma unroll
    for (int et = 0; et < 4; ++et) {
#pragma unroll
        for (int c = 0; c < 2; ++c) {
            const float* bsrc = W + (size_t)(et * 16 + ncol) * HD + c * 32;
            v16bf b;
#pragma unroll
            for (int e = 0; e < 16; ++e) b[e] = (bf16)bsrc[amap(e, half)];
            acc[et] = wmma_bf16(a[c], b, acc[et]);
        }
    }

    // C/D layout: lane col = ncol, rows i+8*half
#pragma unroll
    for (int et = 0; et < 4; ++et) {
#pragma unroll
        for (int i = 0; i < 8; ++i) {
            int grr = gr0 + i + 8 * half;
            int nn  = grr >> 14;
            int rr  = grr & 16383;
            int ls  = rr >> 4;
            int hh  = rr & 15;
            int e   = et * 16 + ncol;
            size_t idx = transposed
                ? (((size_t)(nn * NH + hh) * HD + e) * SL + ls)   // (N,H,D,L)
                : (((size_t)(nn * NH + hh) * SL + ls) * HD + e);  // (N,H,L,D)
            Out[idx] = (bf16)acc[et][i];
        }
    }
}

// ---------------------------------------------------------------------------
// Kernel 2: Wo (E,E) fp32 -> WoT (k-major) bf16:  WoT[k*EM+n] = Wo[n*EM+k]
// ---------------------------------------------------------------------------
__global__ __launch_bounds__(256) void wo_transpose_kernel(const float* __restrict__ Wo,
                                                           bf16* __restrict__ WoT) {
    int idx = blockIdx.x * 256 + threadIdx.x;   // 1M elements
    int k  = idx >> 10;
    int nn = idx & 1023;
    WoT[idx] = (bf16)Wo[(size_t)nn * EM + k];
}

// ---------------------------------------------------------------------------
// Kernel 3: flash attention. One wave = (n, h, 16-query tile).
// S=Q*K^T (bf16 WMMA), ALiBi + mask + 1/sqrt(E), online softmax, O += P*V.
// ---------------------------------------------------------------------------
__global__ __launch_bounds__(256) void attn_kernel(const bf16* __restrict__ Qp,
                                                   const bf16* __restrict__ Kt,
                                                   const bf16* __restrict__ Vp,
                                                   const int* __restrict__ maskp,
                                                   bf16* __restrict__ Attn) {
    __shared__ bf16 plds[8][16][34];            // per-wave P transpose staging

    const int lane = threadIdx.x & 31;
    const int half = lane >> 4;
    const int ncol = lane & 15;
    const int wv   = threadIdx.x >> 5;
    const int w    = blockIdx.x * 8 + wv;       // 16*16*64 = 16384 tiles
    const int qt   = w & 63;
    const int h    = (w >> 6) & 15;
    const int n    = w >> 10;
    const int q0   = qt * 16;

    const bf16* Qb   = Qp + (size_t)(n * NH + h) * SL * HD;  // (L,D)
    const bf16* Ktb  = Kt + (size_t)(n * NH + h) * HD * SL;  // (D,L)
    const bf16* Vb   = Vp + (size_t)(n * NH + h) * SL * HD;  // (L,D)
    const int* mskrw = maskp + n * SL;

    // Q tile in A layout (two 16x32 K-chunks over D=64)
    const bf16* qrow = Qb + (size_t)(q0 + ncol) * HD;
    v16bf qa[2];
#pragma unroll
    for (int c = 0; c < 2; ++c)
#pragma unroll
        for (int e = 0; e < 16; ++e)
            qa[c][e] = qrow[c * 32 + amap(e, half)];

    v8f o[4];
#pragma unroll
    for (int t = 0; t < 4; ++t) o[t] = (v8f){0.f, 0.f, 0.f, 0.f, 0.f, 0.f, 0.f, 0.f};
    float mrow[8], lrow[8];
#pragma unroll
    for (int i = 0; i < 8; ++i) { mrow[i] = -3.0e38f; lrow[i] = 0.f; }

    const float slope = exp2f(-(float)(h + 1));   // 2^(-alpha*(i+1)), alpha=1
    const float scale = 1.0f / 32.0f;             // 1/sqrt(E)
    const float L2E   = 1.44269504f;

    for (int kb = 0; kb < SL; kb += 32) {
        // ---- S (16x32) = Q * K^T : B[k=d][n=krow] = Kt[d][krow] ----
        v8f s[2];
#pragma unroll
        for (int j = 0; j < 2; ++j) s[j] = (v8f){0.f, 0.f, 0.f, 0.f, 0.f, 0.f, 0.f, 0.f};
#pragma unroll
        for (int j = 0; j < 2; ++j)
#pragma unroll
            for (int c = 0; c < 2; ++c) {
                v16bf b;
#pragma unroll
                for (int e = 0; e < 16; ++e)
                    b[e] = Ktb[(size_t)(c * 32 + amap(e, half)) * SL + kb + j * 16 + ncol];
                s[j] = wmma_bf16(qa[c], b, s[j]);
            }

        // ---- ALiBi bias, mask, scale (energy/32 AFTER bias+mask) ----
#pragma unroll
        for (int j = 0; j < 2; ++j) {
            int   tk   = kb + j * 16 + ncol;
            int   mval = mskrw[tk];
            float fk   = (float)tk;
#pragma unroll
            for (int i = 0; i < 8; ++i) {
                float tq = (float)(q0 + i + 8 * half);
                float e  = s[j][i] - slope * fabsf(tq - fk);
                e        = (mval == 0) ? -1.0e4f : e;
                s[j][i]  = e * scale;
            }
        }

        // ---- online softmax: row reductions across 16-lane halves ----
#pragma unroll
        for (int i = 0; i < 8; ++i) {
            float mx = fmaxf(s[0][i], s[1][i]);
            mx = fmaxf(mx, __shfl_xor(mx, 1));
            mx = fmaxf(mx, __shfl_xor(mx, 2));
            mx = fmaxf(mx, __shfl_xor(mx, 4));
            mx = fmaxf(mx, __shfl_xor(mx, 8));
            float mnew = fmaxf(mrow[i], mx);
            float p0 = exp2f((s[0][i] - mnew) * L2E);
            float p1 = exp2f((s[1][i] - mnew) * L2E);
            float rs = p0 + p1;
            rs += __shfl_xor(rs, 1);
            rs += __shfl_xor(rs, 2);
            rs += __shfl_xor(rs, 4);
            rs += __shfl_xor(rs, 8);
            float alpha = exp2f((mrow[i] - mnew) * L2E);
            lrow[i] = lrow[i] * alpha + rs;
            mrow[i] = mnew;
#pragma unroll
            for (int t = 0; t < 4; ++t) o[t][i] *= alpha;
            s[0][i] = p0;
            s[1][i] = p1;
        }

        // ---- P: C layout -> A layout through LDS (wave-private tile) ----
#pragma unroll
        for (int j = 0; j < 2; ++j)
#pragma unroll
            for (int i = 0; i < 8; ++i)
                plds[wv][i + 8 * half][j * 16 + ncol] = (bf16)s[j][i];
        asm volatile("s_wait_dscnt 0" ::: "memory");
        v16bf pa;
#pragma unroll
        for (int e = 0; e < 16; ++e)
            pa[e] = plds[wv][ncol][amap(e, half)];
        asm volatile("s_wait_dscnt 0" ::: "memory");

        // ---- O (16x64) += P * V : B[k=krow][n=d] = V[krow][d] ----
#pragma unroll
        for (int dt = 0; dt < 4; ++dt) {
            v16bf b;
#pragma unroll
            for (int e = 0; e < 16; ++e)
                b[e] = Vb[(size_t)(kb + amap(e, half)) * HD + dt * 16 + ncol];
            o[dt] = wmma_bf16(pa, b, o[dt]);
        }
    }

    // normalize rows and store as (N, L, E) bf16 for the output GEMM
#pragma unroll
    for (int i = 0; i < 8; ++i) lrow[i] = 1.0f / lrow[i];
#pragma unroll
    for (int dt = 0; dt < 4; ++dt)
#pragma unroll
        for (int i = 0; i < 8; ++i) {
            int row = q0 + i + 8 * half;
            int col = h * HD + dt * 16 + ncol;
            Attn[(size_t)(n * SL + row) * EM + col] = (bf16)(o[dt][i] * lrow[i]);
        }
}

// ---------------------------------------------------------------------------
// Kernel 4: output projection  out = Attn @ Wo^T + bo  (M=16384, N=K=1024)
// Block = 16 rows x 512 cols; wave = 16x64 (4 n-tiles).  B slabs (32k x 512n,
// 32 KB) are DMA'd into LDS by the Tensor Data Mover, double-buffered:
// TDM of chunk i+1 overlaps WMMA on chunk i; all 8 waves share the slab.
// ---------------------------------------------------------------------------
__global__ __launch_bounds__(256) void outproj_kernel(const bf16* __restrict__ A,
                                                      const bf16* __restrict__ WoT,
                                                      const float* __restrict__ bo,
                                                      float* __restrict__ Out) {
    extern __shared__ char smemRaw[];
    bf16* bbuf = (bf16*)smemRaw;            // 2 buffers of 32*512 bf16 (64 KB)

    const int lane = threadIdx.x & 31;
    const int half = lane >> 4;
    const int ncol = lane & 15;
    const int wv   = threadIdx.x >> 5;
    const int mt   = blockIdx.x >> 1;       // 1024 row stripes
    const int nh   = blockIdx.x & 1;        // 2 column halves
    const int m0   = mt * 16;
    const int nblk = nh * 512;
    const int n0   = nblk + wv * 64;

    const bf16* arow = A + (size_t)(m0 + ncol) * EM;

    v8f acc[4];
#pragma unroll
    for (int t = 0; t < 4; ++t) acc[t] = (v8f){0.f, 0.f, 0.f, 0.f, 0.f, 0.f, 0.f, 0.f};

    const int NCH = EM / 32;                // 32 k-chunks
    if (wv == 0) {                          // uniform branch; EXEC ignored by TDM
        tdm_load_2d((unsigned)(size_t)(bbuf),
                    WoT + (size_t)0 * EM + nblk, 512, 32, EM);
    }

    for (int ic = 0; ic < NCH; ++ic) {
        const int kb = ic * 32;
        if (wv == 0) {
            if (ic + 1 < NCH) {
                tdm_load_2d((unsigned)(size_t)(bbuf + ((ic + 1) & 1) * (32 * 512)),
                            WoT + (size_t)(kb + 32) * EM + nblk, 512, 32, EM);
                __builtin_amdgcn_s_wait_tensorcnt(1);   // chunk ic landed
            } else {
                __builtin_amdgcn_s_wait_tensorcnt(0);
            }
        }
        __syncthreads();                    // publish slab to all waves

        const bf16* bt = bbuf + (ic & 1) * (32 * 512);

        __builtin_prefetch(arow + kb + 256, 0, 3);      // global_prefetch next A
        v16bf a;
#pragma unroll
        for (int e = 0; e < 16; ++e)
            a[e] = arow[kb + amap(e, half)];
#pragma unroll
        for (int et = 0; et < 4; ++et) {
            v16bf b;
#pragma unroll
            for (int e = 0; e < 16; ++e)
                b[e] = bt[(size_t)amap(e, half) * 512 + wv * 64 + et * 16 + ncol];
            acc[et] = wmma_bf16(a, b, acc[et]);
        }
        __syncthreads();                    // slab consumed before TDM reuses it
    }

#pragma unroll
    for (int et = 0; et < 4; ++et) {
        int   col  = n0 + et * 16 + ncol;
        float bias = bo[col];
#pragma unroll
        for (int i = 0; i < 8; ++i) {
            int row = m0 + i + 8 * half;
            Out[(size_t)row * EM + col] = acc[et][i] + bias;
        }
    }
}

// ---------------------------------------------------------------------------
extern "C" void kernel_launch(void* const* d_in, const int* in_sizes, int n_in,
                              void* d_out, int out_size, void* d_ws, size_t ws_size,
                              hipStream_t stream) {
    (void)in_sizes; (void)n_in; (void)out_size; (void)ws_size;
    const float* values  = (const float*)d_in[0];
    const float* keys    = (const float*)d_in[1];
    const float* queries = (const float*)d_in[2];
    const int*   mask    = (const int*)d_in[3];
    const float* Wv      = (const float*)d_in[4];
    const float* Wk      = (const float*)d_in[5];
    const float* Wq      = (const float*)d_in[6];
    const float* Wo      = (const float*)d_in[7];
    const float* bo      = (const float*)d_in[8];
    float* out = (float*)d_out;

    const size_t projElems = (size_t)NB * NH * SL * HD;   // 16M bf16 each
    bf16* Qp   = (bf16*)d_ws;                 // (N,H,L,D)
    bf16* Kt   = Qp + projElems;              // (N,H,D,L)  transposed
    bf16* Vp   = Kt + projElems;              // (N,H,L,D)
    bf16* Attn = Vp + projElems;              // (N,L,E)
    bf16* WoT  = Attn + (size_t)NB * SL * EM; // (E,E) k-major

    dim3 blk(256);
    proj_kernel<<<2048, blk, 0, stream>>>(queries, Wq, Qp, 0);
    proj_kernel<<<2048, blk, 0, stream>>>(keys,    Wk, Kt, 1);
    proj_kernel<<<2048, blk, 0, stream>>>(values,  Wv, Vp, 0);
    wo_transpose_kernel<<<4096, blk, 0, stream>>>(Wo, WoT);
    attn_kernel<<<2048, blk, 0, stream>>>(Qp, Kt, Vp, mask, Attn);
    outproj_kernel<<<2048, blk, 2 * 32 * 512 * sizeof(bf16), stream>>>(Attn, WoT, bo, out);
}
